// Graph_Layer_Wrapper_44787918963013
// MI455X (gfx1250) — matrix operand order repaired and checked by reading, not compiled, not run on hardware
//
#include <hip/hip_runtime.h>

#define N_ROWS 8192
#define DIM    512
#define DOUT   512
#define KEEP   64
#define CHUNK  2048
#define NCHUNK (N_ROWS / CHUNK)

typedef __attribute__((ext_vector_type(16))) __bf16 v16bf;
typedef __attribute__((ext_vector_type(8)))  __bf16 v8bf;
typedef __attribute__((ext_vector_type(8)))  float  v8f;
typedef __attribute__((ext_vector_type(4)))  unsigned int u32x4;
typedef __attribute__((ext_vector_type(4)))  int i32x4;
typedef __attribute__((ext_vector_type(8)))  int i32x8;

__device__ __forceinline__ unsigned short f32_to_bf16(float f) {
    unsigned u = __builtin_bit_cast(unsigned, f);
    unsigned r = u + 0x7FFFu + ((u >> 16) & 1u);
    return (unsigned short)(r >> 16);
}

#if __has_builtin(__builtin_amdgcn_tensor_load_to_lds)
#define HAVE_TDM 1
// Build a 2D tensor D# (bf16 elements) and issue TENSOR_LOAD_TO_LDS.
__device__ __forceinline__ void tdm_load_2d_bf16(unsigned lds_off,
                                                 const unsigned short* gptr,
                                                 int rows, int cols, int row_stride) {
    unsigned long long ga = (unsigned long long)(uintptr_t)gptr;
    u32x4 g0;
    g0[0] = 1u;                                        // count=1 (valid user D#)
    g0[1] = lds_off;                                   // lds_addr (bytes)
    g0[2] = (unsigned)(ga & 0xFFFFFFFFu);              // global_addr[31:0]
    g0[3] = (unsigned)((ga >> 32) & 0x01FFFFFFu)       // global_addr[56:32]
          | (2u << 30);                                // type=2 ("image")
    i32x8 g1;
    g1[0] = 0x00010000;                                // data_size=1 (2B), mask=0
    g1[1] = (int)((cols & 0xFFFF) << 16);              // tensor_dim0[15:0]
    g1[2] = (int)(((cols >> 16) & 0xFFFF) | ((rows & 0xFFFF) << 16));  // dim0 hi | dim1 lo
    g1[3] = (int)(((rows >> 16) & 0xFFFF) | ((cols & 0xFFFF) << 16));  // dim1 hi | tile_dim0
    g1[4] = (int)(rows & 0xFFFF);                      // tile_dim1 | tile_dim2=0
    g1[5] = row_stride;                                // tensor_dim0_stride[31:0]
    g1[6] = 0;                                         // stride hi | dim1_stride lo
    g1[7] = 0;
    i32x4 z4 = {0, 0, 0, 0};
#if __has_include(<hip/amd_detail/amd_gfx1250_TDM.h>)
    i32x8 z8 = {0, 0, 0, 0, 0, 0, 0, 0};
    __builtin_amdgcn_tensor_load_to_lds(g0, g1, z4, z4, z8, 0);   // clang-23 6-arg
#else
    __builtin_amdgcn_tensor_load_to_lds(g0, g1, z4, z4, 0);       // ROCm 7.2 5-arg
#endif
}
#else
#define HAVE_TDM 0
#endif

// ---------------- Kernel 1: clip + L2-normalize rows -> bf16 ----------------
__global__ void normalize_rows_kernel(const float* __restrict__ sim,
                                      unsigned short* __restrict__ sbf) {
    const int wave = threadIdx.x >> 5;
    const int lane = threadIdx.x & 31;
    const int row  = blockIdx.x * 8 + wave;
    const float* src = sim + (size_t)row * DIM;
    float v[16];
    float ss = 0.f;
#pragma unroll
    for (int t = 0; t < 16; ++t) {
        float f = src[lane + 32 * t];
        f = fminf(1.f, fmaxf(-1.f, f));     // Hardtanh
        v[t] = f;
        ss += f * f;
    }
#pragma unroll
    for (int off = 16; off; off >>= 1) ss += __shfl_xor(ss, off, 32);
    const float inv = 1.f / fmaxf(sqrtf(ss), 1e-12f);
    unsigned short* dst = sbf + (size_t)row * DIM;
#pragma unroll
    for (int t = 0; t < 16; ++t)
        dst[lane + 32 * t] = f32_to_bf16(v[t] * inv);
}

// ---------------- Kernel 2: weight -> transposed bf16 ----------------
__global__ void weight_to_bf16t_kernel(const float* __restrict__ w,
                                       unsigned short* __restrict__ wt) {
    const int idx = blockIdx.x * blockDim.x + threadIdx.x;  // over 512*512
    const int d = idx >> 9;   // row of W
    const int o = idx & 511;  // col of W
    wt[(size_t)o * DIM + d] = f32_to_bf16(w[(size_t)d * DOUT + o]);
}

// ------ Kernel 3: fused  G-tile WMMA -> LDS -> top-64 merge -> (Gs @ x) ------
__global__ void __launch_bounds__(256, 1)
affinity_topk_kernel(const unsigned short* __restrict__ sbf,
                     const float* __restrict__ x,
                     unsigned short* __restrict__ otb) {
    extern __shared__ float chunkbuf[];        // [16][CHUNK] f32 = 128 KB
    __shared__ unsigned short alds[16 * DIM];  // 16 KB A strip (TDM destination)
    __shared__ float topv_s[16][KEEP];
    __shared__ int   topi_s[16][KEEP];
    __shared__ float newv_s[16][KEEP];
    __shared__ int   newi_s[16][KEEP];

    const int wave  = threadIdx.x >> 5;
    const int lane  = threadIdx.x & 31;
    const int ibase = blockIdx.x * 16;
    const int lrow  = lane & 15;
    const int lhalf = lane >> 4;

    // ---- Stage the 16x512 bf16 A strip into LDS (reused by all 512 col tiles)
#if HAVE_TDM
    if (threadIdx.x == 0) {
        unsigned lds_off =
            (unsigned)(uintptr_t)(__attribute__((address_space(3))) unsigned short*)alds;
        tdm_load_2d_bf16(lds_off, sbf + (size_t)ibase * DIM, 16, DIM, DIM);
        __builtin_amdgcn_s_wait_tensorcnt(0);
    }
#else
    {   // cooperative fallback copy: 16 KB / 256 threads = 64 B each
        const uint4* src = (const uint4*)(sbf + (size_t)ibase * DIM);
        uint4* dst = (uint4*)alds;
        for (int t = threadIdx.x; t < 16 * DIM / 8; t += 256) dst[t] = src[t];
    }
#endif
    __syncthreads();

    // ---- Preload all 16 A k-fragments into registers (128 VGPRs)
    v16bf areg[16];
#pragma unroll
    for (int kk = 0; kk < 16; ++kk) {
        const unsigned short* ar = alds + lrow * DIM + kk * 32;
        areg[kk] = *(const v16bf*)(ar + lhalf * 16);
    }

    int oldn = 0;
    for (int c = 0; c < NCHUNK; ++c) {
        // ---- compute phase: 16 x CHUNK strip of G via v_wmma_f32_16x16x32_bf16
        for (int t = wave; t < CHUNK / 16; t += 8) {
            const int jg = c * CHUNK + t * 16;
            {   // prefetch a future B strip (clamped, speculative)
                int jpre = jg + 8 * 16;
                if (jpre > N_ROWS - 16) jpre = N_ROWS - 16;
                __builtin_prefetch(sbf + (size_t)(jpre + lrow) * DIM, 0, 0);
            }
            v8f acc = {};
            const unsigned short* brow = sbf + (size_t)(jg + lrow) * DIM + lhalf * 16;
#pragma unroll
            for (int kk = 0; kk < 16; ++kk) {
                v16bf b = *(const v16bf*)(brow + kk * 32);
                acc = __builtin_amdgcn_wmma_f32_16x16x32_bf16(
                        false, areg[kk], false, b, (short)0, acc, false, false);
            }
            const int coll = t * 16 + lrow;
#pragma unroll
            for (int v = 0; v < 8; ++v) {
                const int m = v + lhalf * 8;
                chunkbuf[(size_t)m * CHUNK + coll] = acc[v];
            }
        }
        __syncthreads();
        // ---- merge phase: each wave owns 2 rows; sorted top-64 union merge
        for (int rr = 0; rr < 2; ++rr) {
            const int r = wave * 2 + rr;
            float* crow = chunkbuf + (size_t)r * CHUNK;
            int po = 0;
            for (int it = 0; it < KEEP; ++it) {
                float bv = -3.402823e38f; int bi = 0;
                for (int tt = lane; tt < CHUNK; tt += 32) {
                    const float f = crow[tt];
                    if (f > bv) { bv = f; bi = tt; }
                }
#pragma unroll
                for (int off = 16; off; off >>= 1) {
                    const float ov = __shfl_xor(bv, off, 32);
                    const int   oi = __shfl_xor(bi, off, 32);
                    if (ov > bv || (ov == bv && oi < bi)) { bv = ov; bi = oi; }
                }
                float nv; int ni;
                if (po < oldn && topv_s[r][po] >= bv) {
                    nv = topv_s[r][po]; ni = topi_s[r][po]; ++po;
                } else {
                    nv = bv; ni = c * CHUNK + bi;
                    if (lane == 0) crow[bi] = -3.402823e38f;
                }
                if (lane == 0) { newv_s[r][it] = nv; newi_s[r][it] = ni; }
            }
            for (int tt = lane; tt < KEEP; tt += 32) {
                topv_s[r][tt] = newv_s[r][tt];
                topi_s[r][tt] = newi_s[r][tt];
            }
        }
        oldn = KEEP;
        __syncthreads();
    }
    // ---- aggregation: out_tmp[row] = (sum_k v_k * x[idx_k]) / sum_k v_k
    for (int rr = 0; rr < 2; ++rr) {
        const int r = wave * 2 + rr;
        float sum = 0.f;
        for (int tt = lane; tt < KEEP; tt += 32) sum += topv_s[r][tt];
#pragma unroll
        for (int off = 16; off; off >>= 1) sum += __shfl_xor(sum, off, 32);
        const float inv = 1.f / sum;
        float acc[16];
#pragma unroll
        for (int t = 0; t < 16; ++t) acc[t] = 0.f;
        for (int kk = 0; kk < KEEP; ++kk) {
            const float w = topv_s[r][kk];
            const float* xr = x + (size_t)topi_s[r][kk] * DIM;
#pragma unroll
            for (int t = 0; t < 16; ++t) acc[t] += w * xr[lane + 32 * t];
        }
        unsigned short* dst = otb + (size_t)(ibase + r) * DIM;
#pragma unroll
        for (int t = 0; t < 16; ++t)
            dst[lane + 32 * t] = f32_to_bf16(acc[t] * inv);
    }
}

// ---------------- Kernel 4: out = out_tmp @ W  (WMMA bf16 -> f32) ----------------
__global__ void __launch_bounds__(256)
out_gemm_kernel(const unsigned short* __restrict__ otb,
                const unsigned short* __restrict__ wt,
                float* __restrict__ out) {
    const int wave  = threadIdx.x >> 5;
    const int lane  = threadIdx.x & 31;
    const int tile  = blockIdx.x * 8 + wave;       // 16384 tiles
    const int ti    = tile >> 5;                   // row tile (512)
    const int tj    = tile & 31;                   // col tile (32)
    const int ibase = ti * 16;
    const int jbase = tj * 16;
    const int lrow  = lane & 15;
    const int lhalf = lane >> 4;
    v8f acc = {};
    const unsigned short* arow = otb + (size_t)(ibase + lrow) * DIM;
    const unsigned short* brow = wt  + (size_t)(jbase + lrow) * DIM + lhalf * 16;
#pragma unroll 4
    for (int k = 0; k < DIM; k += 32) {
        v16bf a = *(const v16bf*)(arow + k + lhalf * 16);
        v16bf b = *(const v16bf*)(brow + k);
        acc = __builtin_amdgcn_wmma_f32_16x16x32_bf16(
                false, a, false, b, (short)0, acc, false, false);
    }
#pragma unroll
    for (int v = 0; v < 8; ++v) {
        const int m = v + lhalf * 8;
        out[(size_t)(ibase + m) * DOUT + jbase + lrow] = acc[v];
    }
}

extern "C" void kernel_launch(void* const* d_in, const int* in_sizes, int n_in,
                              void* d_out, int out_size, void* d_ws, size_t ws_size,
                              hipStream_t stream) {
    const float* x   = (const float*)d_in[0];   // [8192,512]
    const float* sim = (const float*)d_in[1];   // [8192,512]
    const float* w   = (const float*)d_in[2];   // [512,512]
    // d_in[3] = to_keep (==64, hardcoded as KEEP)
    (void)in_sizes; (void)n_in; (void)out_size; (void)ws_size;

    unsigned short* sbf = (unsigned short*)d_ws;                 // 8 MB  bf16 s
    unsigned short* otb = sbf + (size_t)N_ROWS * DIM;            // 8 MB  bf16 Gs@x
    unsigned short* wtb = otb + (size_t)N_ROWS * DIM;            // 0.5MB bf16 W^T

    normalize_rows_kernel<<<N_ROWS / 8, 256, 0, stream>>>(sim, sbf);
    weight_to_bf16t_kernel<<<(DIM * DOUT) / 256, 256, 0, stream>>>(w, wtb);
    affinity_topk_kernel<<<N_ROWS / 16, 256, 16 * CHUNK * sizeof(float), stream>>>(
        sbf, x, otb);
    out_gemm_kernel<<<(N_ROWS / 16) * (DOUT / 16) / 8, 256, 0, stream>>>(
        otb, wtb, (float*)d_out);
}
